// NodeUpdate_58291296141980
// MI455X (gfx1250) — matrix operand order repaired and checked by reading, not compiled, not run on hardware
//
#include <hip/hip_runtime.h>
#include <hip/hip_bf16.h>

// B=256, N=128, D=512, H=512
#define BATCH 256
#define NN 128
#define DD 512
#define HH 512

typedef __attribute__((ext_vector_type(16))) __bf16 v16bf;
typedef __attribute__((ext_vector_type(8)))  float  v8f;

__device__ __forceinline__ unsigned short f2bf(float x) {
    unsigned int u = __float_as_uint(x);
    unsigned int r = u + 0x7FFFu + ((u >> 16) & 1u);   // round-to-nearest-even
    return (unsigned short)(r >> 16);
}
__device__ __forceinline__ unsigned int pack2bf(float a, float b) {
    return (unsigned int)f2bf(a) | ((unsigned int)f2bf(b) << 16);
}

// ---------------------------------------------------------------------------
// Kernel 1a: row-normalize adjacency A (B,N,N) f32 -> bf16. One wave per row.
// ---------------------------------------------------------------------------
__global__ void __launch_bounds__(256) k_norm_adj(const float* __restrict__ A,
                                                  unsigned short* __restrict__ A16) {
    int wave = threadIdx.x >> 5;
    int lane = threadIdx.x & 31;
    long row = (long)blockIdx.x * 8 + wave;            // 0 .. B*N-1
    const float4 a4 = *(const float4*)(A + row * NN + lane * 4);
    float s = a4.x + a4.y + a4.z + a4.w;
    #pragma unroll
    for (int m = 16; m >= 1; m >>= 1) s += __shfl_xor(s, m, 32);
    float inv = 1.0f / (s + 1e-6f);
    uint2 p;
    p.x = pack2bf(a4.x * inv, a4.y * inv);
    p.y = pack2bf(a4.z * inv, a4.w * inv);
    *(uint2*)(A16 + row * NN + lane * 4) = p;
}

// ---------------------------------------------------------------------------
// Kernel 1b: W (H,D) f32 -> bf16 (kept row-major: already K-contiguous per h).
// ---------------------------------------------------------------------------
__global__ void __launch_bounds__(256) k_cvt_w(const float* __restrict__ W,
                                               unsigned int* __restrict__ W16) {
    int idx = blockIdx.x * 256 + threadIdx.x;          // word index (2 elems)
    float2 w2 = *(const float2*)(W + (long)idx * 2);
    W16[idx] = pack2bf(w2.x, w2.y);
}

// ---------------------------------------------------------------------------
// Kernel 1c: X (B,N,D) f32 -> Xt (B,D,N) bf16, LDS-tiled transpose.
// Tile: 32 d-columns x 128 n-rows per block. Pad 130 -> conflict-free.
// ---------------------------------------------------------------------------
__global__ void __launch_bounds__(256) k_transpose_x(const float* __restrict__ X,
                                                     unsigned short* __restrict__ Xt) {
    __shared__ unsigned short tile[32][130];           // [d][n]
    int b  = blockIdx.x >> 4;
    int d0 = (blockIdx.x & 15) * 32;
    const float* Xb = X + (long)b * NN * DD;
    int dl = threadIdx.x & 31;
    int nb = threadIdx.x >> 5;
    #pragma unroll 4
    for (int i = 0; i < 16; ++i) {
        int n = i * 8 + nb;
        tile[dl][n] = f2bf(Xb[(long)n * DD + d0 + dl]);
    }
    __syncthreads();
    unsigned short* XtB = Xt + (long)b * DD * NN;
    #pragma unroll 4
    for (int i = 0; i < 8; ++i) {
        int w  = threadIdx.x + i * 256;                // 0..2047 (32 rows x 64 words)
        int d  = w >> 6;
        int nw = w & 63;
        unsigned int v = (unsigned int)tile[d][2 * nw] |
                         ((unsigned int)tile[d][2 * nw + 1] << 16);
        *(unsigned int*)(XtB + (long)(d0 + d) * NN + 2 * nw) = v;
    }
}

// ---------------------------------------------------------------------------
// Kernel 2: X_agg = X + A_norm @ X   (per batch, 128x128 @ 128x512, bf16 WMMA)
// Block = 8 waves = 32 node-rows x 512 feature-cols. Wave: 16 x 128 strip.
// A-frag from A16 (row-major), B-frag from Xt (feature-major -> K contiguous).
// ---------------------------------------------------------------------------
__global__ void __launch_bounds__(256) k_aggregate(
    const unsigned short* __restrict__ A16,   // (B,128,128) bf16
    const unsigned short* __restrict__ Xt16,  // (B,512,128) bf16
    const float* __restrict__ X,              // (B,128,512) f32
    unsigned short* __restrict__ Xagg) {      // (B,128,512) bf16
    __shared__ float tile[16][516];           // stride 516: rows 8 apart -> bank+32
    int lane = threadIdx.x & 31;
    int w    = threadIdx.x >> 5;
    int mt   = w >> 2;                        // 0..1 : which 16-row half
    int slab = w & 3;                         // 0..3 : which 128-col slab
    int lm = lane & 15, lh = lane >> 4;
    int b        = blockIdx.x >> 2;
    int nodeBase = (blockIdx.x & 3) * 32;

    const unsigned short* Arow  = A16 + ((long)b * NN + nodeBase + mt * 16 + lm) * NN;
    const unsigned short* Bbase = Xt16 + (long)b * DD * NN;

    v8f acc[8];
    #pragma unroll
    for (int i = 0; i < 8; ++i) { v8f z = {}; acc[i] = z; }

    #pragma unroll
    for (int k0 = 0; k0 < NN; k0 += 32) {
        union { v16bf v; uint4 q[2]; } af;
        af.q[0] = *(const uint4*)(Arow + k0 + 8 * lh);        // j=0..7  -> K=k0+8*lh+j
        af.q[1] = *(const uint4*)(Arow + k0 + 16 + 8 * lh);   // j=8..15 -> K=k0+16+8*lh+..
        #pragma unroll
        for (int nt = 0; nt < 8; ++nt) {
            int n = slab * 128 + nt * 16 + lm;                // output feature col
            const unsigned short* bp = Bbase + (long)n * NN + k0 + 8 * lh;
            union { v16bf v; uint4 q[2]; } bfg;
            bfg.q[0] = *(const uint4*)(bp);
            bfg.q[1] = *(const uint4*)(bp + 16);
            acc[nt] = __builtin_amdgcn_wmma_f32_16x16x32_bf16(
                false, af.v, false, bfg.v, (short)0, acc[nt], false, false);
        }
    }

    // Epilogue: acc -> LDS, add X (f32), pack bf16, coalesced store. Two phases.
    const float* Xb = X + (long)b * NN * DD;
    unsigned short* Ob = Xagg + (long)b * NN * DD;
    for (int phase = 0; phase < 2; ++phase) {
        if (mt == phase) {
            #pragma unroll
            for (int nt = 0; nt < 8; ++nt)
                #pragma unroll
                for (int r = 0; r < 8; ++r)
                    tile[r + 8 * lh][slab * 128 + nt * 16 + lm] = acc[nt][r];
        }
        __syncthreads();
        {
            int row = threadIdx.x >> 4;                       // 0..15
            int seg = threadIdx.x & 15;                       // 32 cols each
            int node = nodeBase + phase * 16 + row;
            const float* xr = Xb + (long)node * DD + seg * 32;
            unsigned short* orow = Ob + (long)node * DD + seg * 32;
            #pragma unroll
            for (int i = 0; i < 8; ++i) {
                float4 xv = *(const float4*)(xr + i * 4);
                float v0 = tile[row][seg * 32 + i * 4 + 0] + xv.x;
                float v1 = tile[row][seg * 32 + i * 4 + 1] + xv.y;
                float v2 = tile[row][seg * 32 + i * 4 + 2] + xv.z;
                float v3 = tile[row][seg * 32 + i * 4 + 3] + xv.w;
                uint2 p; p.x = pack2bf(v0, v1); p.y = pack2bf(v2, v3);
                *(uint2*)(orow + i * 4) = p;
            }
        }
        __syncthreads();
    }
}

// ---------------------------------------------------------------------------
// Kernel 3: X_up = Xagg @ W^T + b ; LayerNorm(H) ; sigmoid.  (32768x512x512)
// Block covers 32 token-rows x full H=512 so LN reduces entirely in-block.
// ---------------------------------------------------------------------------
__global__ void __launch_bounds__(256) k_linear_ln(
    const unsigned short* __restrict__ Xagg,  // (32768,512) bf16
    const unsigned short* __restrict__ W16,   // (512,512) bf16 [h][d]
    const float* __restrict__ bias,
    const float* __restrict__ lnw,
    const float* __restrict__ lnb,
    float* __restrict__ out) {                // (32768,512) f32
    __shared__ float tile[16][516];
    int lane = threadIdx.x & 31;
    int w    = threadIdx.x >> 5;
    int mt   = w >> 2;
    int slab = w & 3;
    int lm = lane & 15, lh = lane >> 4;
    long M0 = (long)blockIdx.x * 32;

    const unsigned short* Arow = Xagg + (M0 + mt * 16 + lm) * DD;
    v8f acc[8];
    #pragma unroll
    for (int i = 0; i < 8; ++i) { v8f z = {}; acc[i] = z; }

    for (int k0 = 0; k0 < DD; k0 += 32) {
        union { v16bf v; uint4 q[2]; } af;
        af.q[0] = *(const uint4*)(Arow + k0 + 8 * lh);
        af.q[1] = *(const uint4*)(Arow + k0 + 16 + 8 * lh);
        #pragma unroll
        for (int nt = 0; nt < 8; ++nt) {
            int h = slab * 128 + nt * 16 + lm;
            const unsigned short* bp = W16 + (long)h * DD + k0 + 8 * lh;
            union { v16bf v; uint4 q[2]; } bfg;
            bfg.q[0] = *(const uint4*)(bp);
            bfg.q[1] = *(const uint4*)(bp + 16);
            acc[nt] = __builtin_amdgcn_wmma_f32_16x16x32_bf16(
                false, af.v, false, bfg.v, (short)0, acc[nt], false, false);
        }
    }

    for (int phase = 0; phase < 2; ++phase) {
        if (mt == phase) {
            #pragma unroll
            for (int nt = 0; nt < 8; ++nt)
                #pragma unroll
                for (int r = 0; r < 8; ++r)
                    tile[r + 8 * lh][slab * 128 + nt * 16 + lm] = acc[nt][r];
        }
        __syncthreads();
        {
            int row = threadIdx.x >> 4;                 // 0..15
            int seg = threadIdx.x & 15;                 // 32 cols each
            float vals[32];
            float s = 0.0f, ss = 0.0f;
            #pragma unroll
            for (int i = 0; i < 8; ++i) {
                float4 bv = *(const float4*)(bias + seg * 32 + i * 4);
                float v0 = tile[row][seg * 32 + i * 4 + 0] + bv.x;
                float v1 = tile[row][seg * 32 + i * 4 + 1] + bv.y;
                float v2 = tile[row][seg * 32 + i * 4 + 2] + bv.z;
                float v3 = tile[row][seg * 32 + i * 4 + 3] + bv.w;
                vals[i * 4 + 0] = v0; vals[i * 4 + 1] = v1;
                vals[i * 4 + 2] = v2; vals[i * 4 + 3] = v3;
                s  += v0 + v1 + v2 + v3;
                ss += v0 * v0 + v1 * v1 + v2 * v2 + v3 * v3;
            }
            #pragma unroll
            for (int m = 8; m >= 1; m >>= 1) {          // 16-lane (per-row) reduce
                s  += __shfl_xor(s,  m, 32);
                ss += __shfl_xor(ss, m, 32);
            }
            float mean = s * (1.0f / (float)HH);
            float var  = ss * (1.0f / (float)HH) - mean * mean;
            float rstd = rsqrtf(var + 1e-5f);
            long rg = M0 + phase * 16 + row;
            float* op = out + rg * HH + seg * 32;
            #pragma unroll
            for (int i = 0; i < 8; ++i) {
                float4 gw = *(const float4*)(lnw + seg * 32 + i * 4);
                float4 gb = *(const float4*)(lnb + seg * 32 + i * 4);
                float4 o;
                float y0 = (vals[i*4+0] - mean) * rstd * gw.x + gb.x;
                float y1 = (vals[i*4+1] - mean) * rstd * gw.y + gb.y;
                float y2 = (vals[i*4+2] - mean) * rstd * gw.z + gb.z;
                float y3 = (vals[i*4+3] - mean) * rstd * gw.w + gb.w;
                o.x = 1.0f / (1.0f + __expf(-y0));
                o.y = 1.0f / (1.0f + __expf(-y1));
                o.z = 1.0f / (1.0f + __expf(-y2));
                o.w = 1.0f / (1.0f + __expf(-y3));
                *(float4*)(op + i * 4) = o;
            }
        }
        __syncthreads();
    }
}

// ---------------------------------------------------------------------------
extern "C" void kernel_launch(void* const* d_in, const int* in_sizes, int n_in,
                              void* d_out, int out_size, void* d_ws, size_t ws_size,
                              hipStream_t stream) {
    const float* X    = (const float*)d_in[0];
    const float* A    = (const float*)d_in[1];
    const float* W    = (const float*)d_in[2];
    const float* bias = (const float*)d_in[3];
    const float* lnw  = (const float*)d_in[4];
    const float* lnb  = (const float*)d_in[5];
    float* out = (float*)d_out;

    char* ws = (char*)d_ws;
    // ws layout (bytes):
    //   A16  : B*N*N*2  = 8,388,608
    //   Xt16 : B*D*N*2  = 33,554,432
    //   W16  : H*D*2    =    524,288
    //   Xagg : B*N*D*2  = 33,554,432   (total ~76 MB)
    unsigned short* A16  = (unsigned short*)(ws);
    unsigned short* Xt16 = (unsigned short*)(ws + 8388608);
    unsigned int*   W16  = (unsigned int*)(ws + 8388608 + 33554432);
    unsigned short* Xagg = (unsigned short*)(ws + 8388608 + 33554432 + 524288);

    k_norm_adj   <<<(BATCH * NN) / 8, 256, 0, stream>>>(A, A16);
    k_cvt_w      <<<(HH * DD / 2) / 256, 256, 0, stream>>>(W, W16);
    k_transpose_x<<<BATCH * (DD / 32), 256, 0, stream>>>(X, Xt16);
    k_aggregate  <<<BATCH * (NN / 32), 256, 0, stream>>>(A16, Xt16, X, Xagg);
    k_linear_ln  <<<(BATCH * NN) / 32, 256, 0, stream>>>(
        Xagg, (const unsigned short*)W16, bias, lnw, lnb, out);
}
